// StructuredWindowAttention_2018634629348
// MI455X (gfx1250) — matrix operand order repaired and checked by reading, not compiled, not run on hardware
//
#include <hip/hip_runtime.h>
#include <hip/hip_fp16.h>

typedef __attribute__((ext_vector_type(16))) _Float16 v16h;
typedef __attribute__((ext_vector_type(8)))  _Float16 v8h;
typedef __attribute__((ext_vector_type(8)))  float    v8f;

#define WMMA_F16(A, B, C) \
  __builtin_amdgcn_wmma_f32_16x16x32_f16(false, (A), false, (B), (short)0, (C), false, false)

static __device__ __forceinline__ v16h cat8(v8h lo, v8h hi) {
  return __builtin_shufflevector(lo, hi, 0,1,2,3,4,5,6,7,8,9,10,11,12,13,14,15);
}

// ---------------------------------------------------------------------------
// Attention kernel: one block = 4 waves = 64 queries of one (b, h).
// LDS: Qs[64][32]f16, Ks[256][32]f16, Vt[32][256]f16,
//      Sw[4][16][256]f32, Pw[4][16][256]f16, Sinv[4][16]f32  -> 135,424 B
// ---------------------------------------------------------------------------
__global__ __launch_bounds__(128) void swin_attn_kernel(
    const float* __restrict__ qkv, const int* __restrict__ rpi,
    const float* __restrict__ mask, const float* __restrict__ rpb,
    _Float16* __restrict__ xws)
{
  extern __shared__ char smem[];
  _Float16* Qs   = (_Float16*)smem;               // 64*32
  _Float16* Ks   = Qs + 64 * 32;                  // 256*32
  _Float16* Vt   = Ks + 256 * 32;                 // 32*256
  float*    Sw   = (float*)(Vt + 32 * 256);       // 4*16*256
  _Float16* Pw   = (_Float16*)(Sw + 4 * 16 * 256);// 4*16*256
  float*    Sinv = (float*)(Pw + 4 * 16 * 256);   // 4*16

  const int tid  = threadIdx.x;
  const int b    = blockIdx.z;
  const int h    = blockIdx.y;
  const int qblk = blockIdx.x;
  const float scale = 0.17677669529663687f;       // hd^-0.5, hd=32

  const size_t tok_base = (size_t)b * 256;

  // Stage K (row-major f16) and V (transposed f16)
  for (int i = tid; i < 2048; i += 128) {
    const int t  = i >> 3;
    const int d4 = (i & 7) * 4;
    const float* src = qkv + (tok_base + t) * 576 + h * 32 + d4;
    const float4 kv = *(const float4*)(src + 192);
    Ks[t * 32 + d4 + 0] = (_Float16)kv.x;
    Ks[t * 32 + d4 + 1] = (_Float16)kv.y;
    Ks[t * 32 + d4 + 2] = (_Float16)kv.z;
    Ks[t * 32 + d4 + 3] = (_Float16)kv.w;
    const float4 vv = *(const float4*)(src + 384);
    Vt[(d4 + 0) * 256 + t] = (_Float16)vv.x;
    Vt[(d4 + 1) * 256 + t] = (_Float16)vv.y;
    Vt[(d4 + 2) * 256 + t] = (_Float16)vv.z;
    Vt[(d4 + 3) * 256 + t] = (_Float16)vv.w;
  }
  // Stage Q (pre-scaled)
  for (int i = tid; i < 512; i += 128) {
    const int r  = i >> 3;
    const int d4 = (i & 7) * 4;
    const float4 qv = *(const float4*)(qkv + (tok_base + qblk * 64 + r) * 576 + h * 32 + d4);
    Qs[r * 32 + d4 + 0] = (_Float16)(qv.x * scale);
    Qs[r * 32 + d4 + 1] = (_Float16)(qv.y * scale);
    Qs[r * 32 + d4 + 2] = (_Float16)(qv.z * scale);
    Qs[r * 32 + d4 + 3] = (_Float16)(qv.w * scale);
  }
  __syncthreads();

  const int wave = tid >> 5;
  const int lane = tid & 31;
  const int hf   = lane >> 4;   // half-wave select
  const int ln   = lane & 15;
  const int q0   = qblk * 64 + wave * 16;

  float*    S = Sw + wave * 16 * 256;
  _Float16* P = Pw + wave * 16 * 256;

  // ---- S = Q K^T : one WMMA per 16x16 tile (K=32 covers hd fully) ----
  const v8h qlo = *(const v8h*)(Qs + (wave * 16 + ln) * 32 + hf * 16);
  const v8h qhi = *(const v8h*)(Qs + (wave * 16 + ln) * 32 + hf * 16 + 8);
  const v16h aQ = cat8(qlo, qhi);

#pragma unroll
  for (int t = 0; t < 16; ++t) {
    const v16h bK = *(const v16h*)(Ks + (t * 16 + ln) * 32 + hf * 16);
    v8f acc = {};
    acc = WMMA_F16(aQ, bK, acc);
#pragma unroll
    for (int r = 0; r < 8; ++r)
      S[(r + 8 * hf) * 256 + t * 16 + ln] = acc[r];
  }

  // ---- bias + mask + softmax (lane owns row ln, columns hf*128..+127) ----
  {
    const int qt = q0 + ln;
    const int cbase = hf * 128;
    const float* mrow = mask + ((size_t)(b & 15) * 256 + qt) * 256;
    const int*   irow = rpi + (size_t)qt * 256;
    float mx = -1e30f;
    for (int j = 0; j < 128; ++j) {
      const int c = cbase + j;
      const float v = S[ln * 256 + c] + rpb[irow[c] * 6 + h] + mrow[c];
      S[ln * 256 + c] = v;
      mx = fmaxf(mx, v);
    }
    mx = fmaxf(mx, __shfl_xor(mx, 16, 32));
    float sum = 0.0f;
    for (int j = 0; j < 128; ++j) {
      const int c = cbase + j;
      const float e = __expf(S[ln * 256 + c] - mx);
      sum += e;
      P[ln * 256 + c] = (_Float16)e;
    }
    sum += __shfl_xor(sum, 16, 32);
    if (hf == 0) Sinv[wave * 16 + ln] = 1.0f / sum;
  }

  // ---- O = P V : 8 k-steps x 2 column tiles ----
  v8f o0 = {}, o1 = {};
#pragma unroll
  for (int k0 = 0; k0 < 256; k0 += 32) {
    const v8h plo = *(const v8h*)(P + ln * 256 + k0 + hf * 16);
    const v8h phi = *(const v8h*)(P + ln * 256 + k0 + hf * 16 + 8);
    const v16h aP = cat8(plo, phi);
    const v16h b0 = *(const v16h*)(Vt + ln * 256 + k0 + hf * 16);
    const v16h b1 = *(const v16h*)(Vt + (16 + ln) * 256 + k0 + hf * 16);
    o0 = WMMA_F16(aP, b0, o0);
    o1 = WMMA_F16(aP, b1, o1);
  }

  // ---- write O (f16) into X workspace, applying 1/rowsum ----
#pragma unroll
  for (int r = 0; r < 8; ++r) {
    const int m = r + 8 * hf;
    const float rs = Sinv[wave * 16 + m];
    const size_t row = tok_base + q0 + m;
    xws[row * 192 + h * 32 + ln]      = (_Float16)(o0[r] * rs);
    xws[row * 192 + h * 32 + 16 + ln] = (_Float16)(o1[r] * rs);
  }
}

// ---------------------------------------------------------------------------
// Convert proj_w (192x192 f32) to f16 once.
// ---------------------------------------------------------------------------
__global__ __launch_bounds__(256) void wcvt_kernel(const float* __restrict__ w,
                                                   _Float16* __restrict__ w16)
{
  const int i = blockIdx.x * 256 + threadIdx.x;
  if (i < 192 * 192) w16[i] = (_Float16)w[i];
}

// ---------------------------------------------------------------------------
// Projection: Y = X @ W^T + pb.  X: 65536x192 f16 (workspace), W16 f16 in L2.
// Block = 256 threads (8 waves x 16 rows), 12 N-tiles, K loop 6 x 32.
// ---------------------------------------------------------------------------
__global__ __launch_bounds__(256) void swin_proj_kernel(
    const _Float16* __restrict__ x, const _Float16* __restrict__ w16,
    const float* __restrict__ pb, float* __restrict__ out)
{
  extern __shared__ char smem[];
  _Float16* Xs = (_Float16*)smem;                 // 128*192 f16 = 48 KB

  const int tid = threadIdx.x;
  const size_t row0 = (size_t)blockIdx.x * 128;

  for (int i = tid; i < 3072; i += 256)
    *(uint4*)(Xs + i * 8) = *(const uint4*)(x + row0 * 192 + i * 8);
  __syncthreads();

  const int wave = tid >> 5;
  const int lane = tid & 31;
  const int hf   = lane >> 4;
  const int ln   = lane & 15;
  const int r0   = wave * 16;

  v8f acc[12] = {};

#pragma unroll
  for (int k0 = 0; k0 < 192; k0 += 32) {
    const v8h xlo = *(const v8h*)(Xs + (r0 + ln) * 192 + k0 + hf * 16);
    const v8h xhi = *(const v8h*)(Xs + (r0 + ln) * 192 + k0 + hf * 16 + 8);
    const v16h aX = cat8(xlo, xhi);
#pragma unroll
    for (int ct = 0; ct < 12; ++ct) {
      const v16h bW = *(const v16h*)(w16 + (ct * 16 + ln) * 192 + k0 + hf * 16);
      acc[ct] = WMMA_F16(aX, bW, acc[ct]);
    }
  }

#pragma unroll
  for (int ct = 0; ct < 12; ++ct) {
    const int col = ct * 16 + ln;
    const float bias = pb[col];
#pragma unroll
    for (int r = 0; r < 8; ++r) {
      const size_t row = row0 + r0 + r + 8 * hf;
      out[row * 192 + col] = acc[ct][r] + bias;
    }
  }
}

extern "C" void kernel_launch(void* const* d_in, const int* in_sizes, int n_in,
                              void* d_out, int out_size, void* d_ws, size_t ws_size,
                              hipStream_t stream) {
  const float* qkv  = (const float*)d_in[0];
  const int*   rpi  = (const int*)d_in[1];
  const float* mask = (const float*)d_in[2];
  const float* rpb  = (const float*)d_in[3];
  const float* pw   = (const float*)d_in[4];
  const float* pb   = (const float*)d_in[5];
  float* out = (float*)d_out;

  _Float16* xws = (_Float16*)d_ws;                                   // 65536*192 f16
  _Float16* w16 = (_Float16*)((char*)d_ws + (size_t)65536 * 192 * 2);// 192*192 f16

  const size_t attn_lds = (64 * 32 + 256 * 32 + 32 * 256) * sizeof(_Float16)
                        + (4 * 16 * 256) * sizeof(float)
                        + (4 * 16 * 256) * sizeof(_Float16)
                        + (4 * 16) * sizeof(float);                  // 135,424 B

  dim3 agrid(4, 6, 256);
  swin_attn_kernel<<<agrid, 128, attn_lds, stream>>>(qkv, rpi, mask, rpb, xws);
  wcvt_kernel<<<(192 * 192 + 255) / 256, 256, 0, stream>>>(pw, w16);
  swin_proj_kernel<<<512, 256, 128 * 192 * sizeof(_Float16), stream>>>(xws, w16, pb, out);
}
